// EvoformerIteration_59476707115315
// MI455X (gfx1250) — compile-verified, matrix operand order checked by reading, not run on hardware
//
#include <hip/hip_runtime.h>
#include <hip/hip_bf16.h>
#include <cstdint>

// ============================================================================
// Evoformer iteration for MI455X (gfx1250, wave32, WMMA bf16 16x16x32).
// All contractions go through v_wmma_f32_16x16x32_bf16; fp32 elsewhere.
// Data movement: global_load_async_to_lds_b128 with double-buffered LDS so
// the async copies overlap the WMMA pipe; fragments are 2x (ds_load_b128).
// ============================================================================

typedef __attribute__((ext_vector_type(16))) __bf16 v16bf;
typedef __attribute__((ext_vector_type(8)))  __bf16 v8bf;
typedef __attribute__((ext_vector_type(8)))  float  v8f;

#define DEVINL __device__ __forceinline__

// K index held by fragment element j for a lane in half `half` (ISA 7.12.2,
// 16-bit A 16x32 layout; B uses the same K ordering with lane = column).
// Elements j=0..7 -> K = base..base+7 (contiguous), j=8..15 -> base+16..+23.
DEVINL int frag_k(int j, int half) {
  int base = half * 8;
  return (j < 8) ? (base + j) : (16 + base + (j - 8));
}

// Build a 16-element fragment from two contiguous 8-element (128-bit) loads.
#define FRAG2(dst, p0, p1)                                   \
  do {                                                       \
    v8bf _lo = *(const v8bf*)(p0);                           \
    v8bf _hi = *(const v8bf*)(p1);                           \
    _Pragma("unroll")                                        \
    for (int _j = 0; _j < 8; ++_j) {                         \
      dst[_j] = _lo[_j];                                     \
      dst[_j + 8] = _hi[_j];                                 \
    }                                                        \
  } while (0)

// Async 16-byte global -> LDS copy (CDNA5 async path, tracked by ASYNCcnt).
DEVINL void async_b128(unsigned lds_addr, const __bf16* gaddr) {
  asm volatile("global_load_async_to_lds_b128 %0, %1, off"
               :: "v"(lds_addr), "v"(gaddr)
               : "memory");
}
DEVINL void wait_async0() {
  asm volatile("s_wait_asynccnt 0" ::: "memory");
}

// ----------------------------------------------------------------------------
// Generic batched bf16 GEMM: C = act(alpha * A*B^(T?) + bias) [+C]
// A: [M,K] (lda, batch stride sA), B: [K,N] or [N,K] if transB (ldb, sB)
// Cout fp32 and/or Cbf bf16 (ldc, sC). Block: 256 thr = 8 waves -> 64x32 tile.
// B is kept transposed in LDS (BsT[n][k]) so fragments are contiguous in K.
// Fast path: double-buffered LDS, async copies issued before the WMMA of the
// previous tile so global latency hides behind the matrix pipe.
// ----------------------------------------------------------------------------
__global__ void __launch_bounds__(256)
evo_gemm_kernel(const __bf16* __restrict__ A, long long sA, int lda,
                const __bf16* __restrict__ B, long long sB, int ldb, int transB,
                float* __restrict__ Cout, __bf16* __restrict__ Cbf,
                long long sC, int ldc,
                const float* __restrict__ bias,
                int M, int N, int K, float alpha, int act, int addC)
{
  __shared__ __align__(16) __bf16 As[2][64][40];
  __shared__ __align__(16) __bf16 BsT[2][32][40];
  const int tid  = threadIdx.x;
  const int lane = tid & 31;
  const int wave = tid >> 5;
  const int half = lane >> 4;
  const int l16  = lane & 15;
  const int base = half * 8;
  const int m0 = blockIdx.y * 64;
  const int n0 = blockIdx.x * 32;
  const int bz = blockIdx.z;
  const __bf16* Ab = A + (size_t)bz * (size_t)sA;
  const __bf16* Bb = B + (size_t)bz * (size_t)sB;
  const int mrow = (wave & 3) * 16;
  const int ncol = (wave >> 2) * 16;
  const bool full = (m0 + 64 <= M) && (n0 + 32 <= N) && ((K & 31) == 0);

  v8f acc = {};
  if (full) {
    // Per-thread staging coordinates.
    const int ar = tid >> 2, ac8 = (tid & 3) * 8;        // A: 64 rows x 4 chunks
    const int bn = tid >> 2, bk8 = (tid & 3) * 8;        // B^T: 32 rows x 4 chunks
    const int bk = tid >> 2, bn8 = (tid & 3) * 8;        // B:   32 rows x 4 chunks

    auto stage_async = [&](int buf, int k0) {
      async_b128((unsigned)(uintptr_t)&As[buf][ar][ac8],
                 Ab + (size_t)(m0 + ar) * lda + (k0 + ac8));
      if (transB && tid < 128)
        async_b128((unsigned)(uintptr_t)&BsT[buf][bn][bk8],
                   Bb + (size_t)(n0 + bn) * ldb + (k0 + bk8));
    };
    auto loadB_sync = [&](int k0) -> v8bf {
      return *(const v8bf*)(Bb + (size_t)(k0 + bk) * ldb + (n0 + bn8));
    };
    auto scatterB = [&](int buf, v8bf v) {
#pragma unroll
      for (int j = 0; j < 8; ++j) BsT[buf][bn8 + j][bk] = v[j];
    };

    const int nt = K >> 5;
    // Prologue: stage tile 0.
    stage_async(0, 0);
    if (!transB && tid < 128) scatterB(0, loadB_sync(0));
    wait_async0();
    __syncthreads();

    v8bf bpend = {};
    for (int i = 0; i < nt; ++i) {
      const int cur = i & 1;
      const bool more = (i + 1) < nt;
      const int k0n = (i + 1) << 5;
      // Issue next tile's copies before this tile's WMMA (latency hiding).
      if (more) {
        stage_async(cur ^ 1, k0n);
        if (!transB && tid < 128) bpend = loadB_sync(k0n);
      }
      v16bf a, b;
      FRAG2(a, &As[cur][mrow + l16][base], &As[cur][mrow + l16][16 + base]);
      FRAG2(b, &BsT[cur][ncol + l16][base], &BsT[cur][ncol + l16][16 + base]);
      acc = __builtin_amdgcn_wmma_f32_16x16x32_bf16(false, a, false, b,
                                                    (short)0, acc, false, false);
      if (more) {
        if (!transB && tid < 128) scatterB(cur ^ 1, bpend);
        wait_async0();
      }
      __syncthreads();
    }
  } else {
    // Guarded edge path (only the tiny N=8 / N=4 bias projections hit this).
    for (int k0 = 0; k0 < K; k0 += 32) {
      for (int i = tid; i < 64 * 32; i += 256) {
        int r = i >> 5, c = i & 31;
        int gm = m0 + r, gk = k0 + c;
        As[0][r][c] = (gm < M && gk < K) ? Ab[(size_t)gm * lda + gk] : (__bf16)0.0f;
      }
      for (int i = tid; i < 32 * 32; i += 256) {
        int kk = i >> 5, nn = i & 31;
        int gk = k0 + kk, gn = n0 + nn;
        __bf16 v = (__bf16)0.0f;
        if (gk < K && gn < N)
          v = transB ? Bb[(size_t)gn * ldb + gk] : Bb[(size_t)gk * ldb + gn];
        BsT[0][nn][kk] = v;
      }
      if (k0 + 32 < K)
        __builtin_prefetch(Ab + (size_t)(m0 + (tid >> 2)) * lda + (k0 + 32), 0, 1);
      __syncthreads();
      v16bf a, b;
      FRAG2(a, &As[0][mrow + l16][base], &As[0][mrow + l16][16 + base]);
      FRAG2(b, &BsT[0][ncol + l16][base], &BsT[0][ncol + l16][16 + base]);
      acc = __builtin_amdgcn_wmma_f32_16x16x32_bf16(false, a, false, b,
                                                    (short)0, acc, false, false);
      __syncthreads();
    }
  }
#pragma unroll
  for (int r = 0; r < 8; ++r) {
    int gm = m0 + mrow + r + half * 8;
    int gn = n0 + ncol + l16;
    if (gm < M && gn < N) {
      float v = acc[r] * alpha + (bias ? bias[gn] : 0.0f);
      if (act == 1) v = fmaxf(v, 0.0f);
      else if (act == 2) v = 1.0f / (1.0f + __expf(-v));
      size_t idx = (size_t)bz * (size_t)sC + (size_t)gm * ldc + gn;
      if (Cout) { if (addC) Cout[idx] += v; else Cout[idx] = v; }
      if (Cbf) Cbf[idx] = (__bf16)v;
    }
  }
}

// ----------------------------------------------------------------------------
// Fused attention: one wave per (b, h, 16-row q tile). D = 32.
// logits = q.k^T (+bias, mask) -> softmax in LDS -> PV, all via WMMA bf16.
// V is staged transposed into LDS so every PV fragment is 2x ds_load_b128.
// mask value = mask[b*mbs + k*mks] (keep if > 0.5).
// bias layout: [(q*Kn + k)*H + h] or null.
// ----------------------------------------------------------------------------
__global__ void __launch_bounds__(32)
evo_attn_kernel(const __bf16* __restrict__ Qm, const __bf16* __restrict__ Km,
                const __bf16* __restrict__ Vm,
                const float* __restrict__ bias, const float* __restrict__ mask,
                int mbs, int mks,
                float* __restrict__ out,
                int B, int Q, int Kn, int H, int HD)
{
  __shared__ __align__(16) float  slog[16][256];
  __shared__ __align__(16) __bf16 sw[16][256];
  __shared__ __align__(16) __bf16 svt[32][264];   // V transposed: [d][k]
  const int lane = threadIdx.x & 31;
  const int half = lane >> 4;
  const int l16  = lane & 15;
  const int base = half * 8;
  const int q0 = blockIdx.x * 16;
  const int h  = blockIdx.y;
  const int b  = blockIdx.z;
  const int D  = 32;

  // Stage V transposed: lane handles row k = r + lane, 32 d values (4x b128).
  const __bf16* vbase = Vm + (size_t)b * Kn * HD + h * D;
  for (int r = 0; r < Kn; r += 32) {
    int k = r + lane;
    const __bf16* vp = vbase + (size_t)k * HD;
    v8bf x0 = *(const v8bf*)(vp);
    v8bf x1 = *(const v8bf*)(vp + 8);
    v8bf x2 = *(const v8bf*)(vp + 16);
    v8bf x3 = *(const v8bf*)(vp + 24);
#pragma unroll
    for (int j = 0; j < 8; ++j) {
      svt[j][k]      = x0[j];
      svt[8 + j][k]  = x1[j];
      svt[16 + j][k] = x2[j];
      svt[24 + j][k] = x3[j];
    }
  }

  // Q fragment: two contiguous 128-bit global loads per lane.
  v16bf qa;
  const __bf16* qbase = Qm + ((size_t)b * Q + q0) * HD + h * D;
  {
    const __bf16* qp = qbase + (size_t)l16 * HD;
    FRAG2(qa, qp + base, qp + 16 + base);
  }

  const __bf16* kbase = Km + (size_t)b * Kn * HD + h * D;
  for (int kt = 0; kt < Kn; kt += 16) {
    v16bf kb;
    const __bf16* kp = kbase + (size_t)(kt + l16) * HD;
    FRAG2(kb, kp + base, kp + 16 + base);
    v8f c = {};
    c = __builtin_amdgcn_wmma_f32_16x16x32_bf16(false, qa, false, kb,
                                                (short)0, c, false, false);
#pragma unroll
    for (int r = 0; r < 8; ++r) {
      int m = r + half * 8;
      int kk = kt + l16;
      float v = c[r];
      if (bias) v += bias[((size_t)(q0 + m) * Kn + kk) * H + h];
      if (mask[(size_t)b * mbs + (size_t)kk * mks] <= 0.5f) v = -1e9f;
      slog[m][kk] = v;
    }
  }
  __syncthreads();
  {
    int row = l16;
    int kh = Kn >> 1;
    int st = half * kh;
    float mx = -1e30f;
    for (int kk = 0; kk < kh; ++kk) mx = fmaxf(mx, slog[row][st + kk]);
    mx = fmaxf(mx, __shfl_xor(mx, 16, 32));
    float sum = 0.0f;
    for (int kk = 0; kk < kh; ++kk) {
      float e = __expf(slog[row][st + kk] - mx);
      slog[row][st + kk] = e; sum += e;
    }
    sum += __shfl_xor(sum, 16, 32);
    float inv = 1.0f / sum;
    for (int kk = 0; kk < kh; ++kk)
      sw[row][st + kk] = (__bf16)(slog[row][st + kk] * inv);
  }
  __syncthreads();
  v8f a0 = {}, a1 = {};
  for (int kc = 0; kc < Kn; kc += 32) {
    v16bf wa, vb0, vb1;
    FRAG2(wa,  &sw[l16][kc + base],        &sw[l16][kc + 16 + base]);
    FRAG2(vb0, &svt[l16][kc + base],       &svt[l16][kc + 16 + base]);
    FRAG2(vb1, &svt[16 + l16][kc + base],  &svt[16 + l16][kc + 16 + base]);
    a0 = __builtin_amdgcn_wmma_f32_16x16x32_bf16(false, wa, false, vb0,
                                                 (short)0, a0, false, false);
    a1 = __builtin_amdgcn_wmma_f32_16x16x32_bf16(false, wa, false, vb1,
                                                 (short)0, a1, false, false);
  }
  float* obase = out + ((size_t)b * Q + q0) * HD + h * D;
#pragma unroll
  for (int r = 0; r < 8; ++r) {
    int m = r + half * 8;
    obase[(size_t)m * HD + l16]      = a0[r];
    obase[(size_t)m * HD + 16 + l16] = a1[r];
  }
}

// ----------------------------------------------------------------------------
// LayerNorm over last dim C (blockDim == C, one row per block).
// ----------------------------------------------------------------------------
__global__ void evo_ln_kernel(const float* __restrict__ x,
                              const float* __restrict__ s,
                              const float* __restrict__ o,
                              float* __restrict__ yf, __bf16* __restrict__ yb,
                              int C)
{
  __shared__ float red[256];
  __shared__ float stat[2];
  int row = blockIdx.x;
  int t = threadIdx.x;
  float v = x[(size_t)row * C + t];
  red[t] = v; __syncthreads();
  for (int st = blockDim.x >> 1; st > 0; st >>= 1) {
    if (t < st) red[t] += red[t + st];
    __syncthreads();
  }
  if (t == 0) stat[0] = red[0] / C;
  __syncthreads();
  float d = v - stat[0];
  red[t] = d * d; __syncthreads();
  for (int st = blockDim.x >> 1; st > 0; st >>= 1) {
    if (t < st) red[t] += red[t + st];
    __syncthreads();
  }
  if (t == 0) stat[1] = rsqrtf(red[0] / C + 1e-5f);
  __syncthreads();
  float y = d * stat[1] * s[t] + o[t];
  if (yf) yf[(size_t)row * C + t] = y;
  if (yb) yb[(size_t)row * C + t] = (__bf16)y;
}

// ---------------------------- elementwise kernels ---------------------------
__global__ void evo_cvt_kernel(const float* __restrict__ in,
                               __bf16* __restrict__ out, size_t n) {
  size_t i = (size_t)blockIdx.x * 256 + threadIdx.x;
  if (i < n) out[i] = (__bf16)in[i];
}

__global__ void evo_mask_rows_kernel(float* __restrict__ x,
                                     const float* __restrict__ m,
                                     __bf16* __restrict__ xb, int C, size_t n) {
  size_t i = (size_t)blockIdx.x * 256 + threadIdx.x;
  if (i < n) {
    float v = x[i] * m[i / C];
    x[i] = v;
    if (xb) xb[i] = (__bf16)v;
  }
}

__global__ void evo_mul_gate_kernel(const float* __restrict__ a,
                                    const __bf16* __restrict__ g,
                                    __bf16* __restrict__ out, size_t n) {
  size_t i = (size_t)blockIdx.x * 256 + threadIdx.x;
  if (i < n) out[i] = (__bf16)(a[i] * (float)g[i]);
}

__global__ void evo_mul_add_res_kernel(float* __restrict__ dst,
                                       const float* __restrict__ a,
                                       const float* __restrict__ b, size_t n) {
  size_t i = (size_t)blockIdx.x * 256 + threadIdx.x;
  if (i < n) dst[i] += a[i] * b[i];
}

// out [B,A,C] = in [A,B,C] transposed over first two dims
__global__ void evo_transpose3_kernel(const float* __restrict__ in,
                                      float* __restrict__ out,
                                      int A, int B, int C, size_t n) {
  size_t idx = (size_t)blockIdx.x * 256 + threadIdx.x;
  if (idx < n) {
    int c = (int)(idx % C);
    size_t t = idx / C;
    int j = (int)(t % B);
    int i = (int)(t / B);
    out[((size_t)j * A + i) * C + c] = in[idx];
  }
}

// dst [A,B,C] += src [B,A,C]
__global__ void evo_add_transpose_kernel(float* __restrict__ dst,
                                         const float* __restrict__ src,
                                         int A, int B, int C, size_t n) {
  size_t idx = (size_t)blockIdx.x * 256 + threadIdx.x;
  if (idx < n) {
    int c = (int)(idx % C);
    size_t t = idx / C;
    int j = (int)(t % B);
    int i = (int)(t / B);
    dst[idx] += src[((size_t)j * A + i) * C + c];
  }
}

// OPM: out_bf[(q*32+c)*S + s] = in[(s*R+q)*32 + c]
__global__ void evo_opm_transpose_kernel(const float* __restrict__ in,
                                         __bf16* __restrict__ out,
                                         int S, int R, size_t n) {
  size_t idx = (size_t)blockIdx.x * 256 + threadIdx.x;
  if (idx < n) {
    int c = (int)(idx & 31);
    size_t t = idx >> 5;
    int q = (int)(t % R);
    int s = (int)(t / R);
    out[((size_t)q * 32 + c) * S + s] = (__bf16)in[idx];
  }
}

// combine mask*g1*sigmoid(g2) and channel-transpose to [C,R,R] bf16
__global__ void evo_chan_ct_kernel(const float* __restrict__ g1,
                                   const float* __restrict__ g2,
                                   const float* __restrict__ m2,
                                   __bf16* __restrict__ out,
                                   int R, int C, int swap, size_t n) {
  size_t idx = (size_t)blockIdx.x * 256 + threadIdx.x;
  if (idx < n) {
    int c = (int)(idx % C);
    size_t r = idx / C;
    int y = (int)(r % R);
    int x = (int)(r / R);
    float v = m2[r] * g1[idx] * g2[idx];
    size_t dst = (size_t)c * R * R + (swap ? ((size_t)y * R + x) : ((size_t)x * R + y));
    out[dst] = (__bf16)v;
  }
}

// out [R*R, C] f32 from in [C, R*R] f32
__global__ void evo_chan_ut_kernel(const float* __restrict__ in,
                                   float* __restrict__ out, int R, int C, size_t n) {
  size_t idx = (size_t)blockIdx.x * 256 + threadIdx.x;
  if (idx < n) {
    int c = (int)(idx % C);
    size_t r = idx / C;
    out[idx] = in[(size_t)c * R * R + r];
  }
}

__global__ void evo_opm_add_kernel(float* __restrict__ pair,
                                   const float* __restrict__ opm,
                                   const float* __restrict__ ob,
                                   const float* __restrict__ norm,
                                   int CZ, size_t n) {
  size_t idx = (size_t)blockIdx.x * 256 + threadIdx.x;
  if (idx < n) {
    int f = (int)(idx % CZ);
    size_t pq = idx / CZ;
    pair[idx] += (opm[idx] + ob[f]) / (1e-3f + norm[pq]);
  }
}

__global__ void evo_mask2bf_kernel(const float* __restrict__ m,
                                   __bf16* __restrict__ mb,
                                   __bf16* __restrict__ mtb, int S, int R) {
  size_t idx = (size_t)blockIdx.x * 256 + threadIdx.x;
  if (idx < (size_t)S * R) {
    int r = (int)(idx % R);
    int s = (int)(idx / R);
    float v = m[idx];
    mb[idx] = (__bf16)v;
    mtb[(size_t)r * S + s] = (__bf16)v;
  }
}

// ============================================================================
// Host orchestration
// ============================================================================
extern "C" void kernel_launch(void* const* d_in, const int* in_sizes, int n_in,
                              void* d_out, int out_size, void* d_ws, size_t ws_size,
                              hipStream_t stream) {
  (void)in_sizes; (void)n_in; (void)out_size; (void)ws_size;
  const int S = 128, R = 256, CM = 256, CZ = 128, HM = 8, HP = 4;
  const int HDM = 256, HDP = 128;
  const float QS = 0.17677669529663687f; // 32^-0.5
  const size_t MSA_N = (size_t)S * R * CM;
  const size_t PAIR_N = (size_t)R * R * CZ;
  const size_t MB = 1ull << 20;
  const size_t WB = 0, B1 = 8 * MB, B2 = 48 * MB, B3 = 120 * MB,
               B4 = 184 * MB, B5 = 248 * MB;

  char* ws = (char*)d_ws;
  auto F32 = [&](size_t off) { return (float*)(ws + off); };
  auto BFP = [&](size_t off) { return (__bf16*)(ws + off); };
  auto in  = [&](int i) { return (const float*)d_in[i]; };
  auto blocks = [](size_t n) { return (unsigned)((n + 255) / 256); };

  float* msa  = (float*)d_out;
  float* pair = msa + MSA_N;
  hipMemcpyAsync(msa, d_in[0], MSA_N * 4, hipMemcpyDeviceToDevice, stream);
  hipMemcpyAsync(pair, d_in[1], PAIR_N * 4, hipMemcpyDeviceToDevice, stream);

  // ---- bf16 weight arena ----
  size_t wb = 0;
  auto cvtw = [&](int idx, size_t n) -> __bf16* {
    __bf16* p = (__bf16*)(ws + WB + wb);
    wb += (n * 2 + 255) & ~(size_t)255;
    evo_cvt_kernel<<<blocks(n), 256, 0, stream>>>((const float*)d_in[idx], p, n);
    return p;
  };
  __bf16* w_row_q = cvtw(9, (size_t)CM * HDM);
  __bf16* w_row_k = cvtw(10, (size_t)CM * HDM);
  __bf16* w_row_v = cvtw(11, (size_t)CM * HDM);
  __bf16* w_row_g = cvtw(12, (size_t)CM * HDM);
  __bf16* w_row_o = cvtw(14, (size_t)HDM * CM);
  __bf16* w_row_f = cvtw(8, (size_t)CZ * HM);
  __bf16* w_col_q = cvtw(18, (size_t)CM * HDM);
  __bf16* w_col_k = cvtw(19, (size_t)CM * HDM);
  __bf16* w_col_v = cvtw(20, (size_t)CM * HDM);
  __bf16* w_col_g = cvtw(21, (size_t)CM * HDM);
  __bf16* w_col_o = cvtw(23, (size_t)HDM * CM);
  __bf16* w_mt1 = cvtw(27, (size_t)CM * 1024);
  __bf16* w_mt2 = cvtw(29, (size_t)1024 * CM);
  __bf16* w_opm_l = cvtw(33, (size_t)CM * 32);
  __bf16* w_opm_r = cvtw(35, (size_t)CM * 32);
  __bf16* w_opm_o = cvtw(37, (size_t)32 * 32 * CZ);
  int triBase[2] = {39, 55};
  __bf16 *tri_lw[2], *tri_lgw[2], *tri_rw[2], *tri_rgw[2], *tri_ow[2], *tri_gw[2];
  for (int t = 0; t < 2; ++t) {
    int b0 = triBase[t];
    tri_lw[t]  = cvtw(b0 + 2, (size_t)CZ * CZ);
    tri_lgw[t] = cvtw(b0 + 4, (size_t)CZ * CZ);
    tri_rw[t]  = cvtw(b0 + 6, (size_t)CZ * CZ);
    tri_rgw[t] = cvtw(b0 + 8, (size_t)CZ * CZ);
    tri_ow[t]  = cvtw(b0 + 12, (size_t)CZ * CZ);
    tri_gw[t]  = cvtw(b0 + 14, (size_t)CZ * CZ);
  }
  int taBase[2] = {71, 81};
  __bf16 *ta_f[2], *ta_q[2], *ta_k[2], *ta_v[2], *ta_g[2], *ta_o[2];
  for (int t = 0; t < 2; ++t) {
    int b0 = taBase[t];
    ta_f[t] = cvtw(b0 + 2, (size_t)CZ * HP);
    ta_q[t] = cvtw(b0 + 3, (size_t)CZ * HDP);
    ta_k[t] = cvtw(b0 + 4, (size_t)CZ * HDP);
    ta_v[t] = cvtw(b0 + 5, (size_t)CZ * HDP);
    ta_g[t] = cvtw(b0 + 6, (size_t)CZ * HDP);
    ta_o[t] = cvtw(b0 + 8, (size_t)HDP * CZ);
  }
  __bf16* w_pt1 = cvtw(93, (size_t)CZ * 512);
  __bf16* w_pt2 = cvtw(95, (size_t)512 * CZ);
  __bf16* mask_bf = (__bf16*)(ws + WB + wb); wb += ((size_t)S * R * 2 + 255) & ~(size_t)255;
  __bf16* maskT_bf = (__bf16*)(ws + WB + wb); wb += ((size_t)S * R * 2 + 255) & ~(size_t)255;
  evo_mask2bf_kernel<<<blocks((size_t)S * R), 256, 0, stream>>>(in(2), mask_bf, maskT_bf, S, R);

  auto gemm = [&](const __bf16* A, long long sA, int lda,
                  const __bf16* B, long long sB, int ldb, int tB,
                  float* C, __bf16* Cb, long long sC, int ldc,
                  const float* bias, int M, int N, int K, int batch,
                  float alpha, int act, int addC) {
    dim3 g((N + 31) / 32, (M + 63) / 64, batch);
    evo_gemm_kernel<<<g, 256, 0, stream>>>(A, sA, lda, B, sB, ldb, tB,
                                           C, Cb, sC, ldc, bias,
                                           M, N, K, alpha, act, addC);
  };
  auto ln = [&](const float* x, int si, int oi, __bf16* yb, float* yf,
                int rows, int C) {
    evo_ln_kernel<<<rows, C, 0, stream>>>(x, in(si), in(oi), yf, yb, C);
  };
  auto attn = [&](const __bf16* q, const __bf16* k, const __bf16* v,
                  const float* bias, const float* mask, int mbs, int mks,
                  float* out, int B, int Q, int Kn, int H, int HD) {
    dim3 g(Q / 16, H, B);
    evo_attn_kernel<<<g, 32, 0, stream>>>(q, k, v, bias, mask, mbs, mks,
                                          out, B, Q, Kn, H, HD);
  };

  __bf16* actb   = BFP(B1);
  __bf16* qb     = BFP(B2);
  __bf16* kb     = BFP(B2 + 16 * MB);
  __bf16* vb     = BFP(B2 + 32 * MB);
  __bf16* gateb  = BFP(B2 + 48 * MB);
  float*  wabuf  = F32(B3);
  __bf16* gatedb = BFP(B4);

  // ===== 1. MSA row attention with pair bias =====
  {
    __bf16* pairnb = BFP(B1 + 16 * MB);
    float* nb = F32(B1 + 32 * MB);
    ln(msa, 4, 5, actb, nullptr, S * R, CM);
    ln(pair, 6, 7, pairnb, nullptr, R * R, CZ);
    gemm(pairnb, 0, CZ, w_row_f, 0, HM, 0, nb, nullptr, 0, HM, nullptr,
         R * R, HM, CZ, 1, 1.0f, 0, 0);
    gemm(actb, 0, CM, w_row_q, 0, HDM, 0, nullptr, qb, 0, HDM, nullptr, S * R, HDM, CM, 1, QS, 0, 0);
    gemm(actb, 0, CM, w_row_k, 0, HDM, 0, nullptr, kb, 0, HDM, nullptr, S * R, HDM, CM, 1, 1.0f, 0, 0);
    gemm(actb, 0, CM, w_row_v, 0, HDM, 0, nullptr, vb, 0, HDM, nullptr, S * R, HDM, CM, 1, 1.0f, 0, 0);
    gemm(actb, 0, CM, w_row_g, 0, HDM, 0, nullptr, gateb, 0, HDM, in(13), S * R, HDM, CM, 1, 1.0f, 2, 0);
    attn(qb, kb, vb, nb, in(2), R, 1, wabuf, S, R, R, HM, HDM);
    evo_mul_gate_kernel<<<blocks(MSA_N), 256, 0, stream>>>(wabuf, gateb, gatedb, MSA_N);
    gemm(gatedb, 0, HDM, w_row_o, 0, CM, 0, msa, nullptr, 0, CM, in(15),
         S * R, CM, HDM, 1, 1.0f, 0, 1);
  }

  // ===== 2. MSA column attention =====
  {
    float* msaT = F32(B5);
    evo_transpose3_kernel<<<blocks(MSA_N), 256, 0, stream>>>(msa, msaT, S, R, CM, MSA_N);
    ln(msaT, 16, 17, actb, nullptr, R * S, CM);
    gemm(actb, 0, CM, w_col_q, 0, HDM, 0, nullptr, qb, 0, HDM, nullptr, R * S, HDM, CM, 1, QS, 0, 0);
    gemm(actb, 0, CM, w_col_k, 0, HDM, 0, nullptr, kb, 0, HDM, nullptr, R * S, HDM, CM, 1, 1.0f, 0, 0);
    gemm(actb, 0, CM, w_col_v, 0, HDM, 0, nullptr, vb, 0, HDM, nullptr, R * S, HDM, CM, 1, 1.0f, 0, 0);
    gemm(actb, 0, CM, w_col_g, 0, HDM, 0, nullptr, gateb, 0, HDM, in(22), R * S, HDM, CM, 1, 1.0f, 2, 0);
    attn(qb, kb, vb, nullptr, in(2), 1, R, wabuf, R, S, S, HM, HDM);
    evo_mul_gate_kernel<<<blocks(MSA_N), 256, 0, stream>>>(wabuf, gateb, gatedb, MSA_N);
    float* colout = F32(B5 + 32 * MB);
    gemm(gatedb, 0, HDM, w_col_o, 0, CM, 0, colout, nullptr, 0, CM, in(24),
         R * S, CM, HDM, 1, 1.0f, 0, 0);
    evo_add_transpose_kernel<<<blocks(MSA_N), 256, 0, stream>>>(msa, colout, S, R, CM, MSA_N);
  }

  // ===== 3. MSA transition =====
  {
    ln(msa, 25, 26, actb, nullptr, S * R, CM);
    __bf16* hid = BFP(B2);
    gemm(actb, 0, CM, w_mt1, 0, 1024, 0, nullptr, hid, 0, 1024, in(28),
         S * R, 1024, CM, 1, 1.0f, 1, 0);
    gemm(hid, 0, 1024, w_mt2, 0, CM, 0, msa, nullptr, 0, CM, in(30),
         S * R, CM, 1024, 1, 1.0f, 0, 1);
  }

  // ===== 4. Outer product mean =====
  {
    ln(msa, 31, 32, actb, nullptr, S * R, CM);
    float* leftf = F32(B3);
    float* rightf = F32(B3 + 4 * MB);
    __bf16* rightb = BFP(B3 + 8 * MB);
    __bf16* L2 = BFP(B3 + 12 * MB);
    float* normf = F32(B3 + 16 * MB);
    gemm(actb, 0, CM, w_opm_l, 0, 32, 0, leftf, nullptr, 0, 32, in(34), S * R, 32, CM, 1, 1.0f, 0, 0);
    gemm(actb, 0, CM, w_opm_r, 0, 32, 0, rightf, nullptr, 0, 32, in(36), S * R, 32, CM, 1, 1.0f, 0, 0);
    size_t nel = (size_t)S * R * 32;
    evo_mask_rows_kernel<<<blocks(nel), 256, 0, stream>>>(leftf, in(2), nullptr, 32, nel);
    evo_mask_rows_kernel<<<blocks(nel), 256, 0, stream>>>(rightf, in(2), rightb, 32, nel);
    evo_opm_transpose_kernel<<<blocks(nel), 256, 0, stream>>>(leftf, L2, S, R, nel);
    __bf16* T = BFP(B5);
    gemm(L2, 0, S, rightb, 32, R * 32, 0, nullptr, T, (long long)R * 32 * 32, 32,
         nullptr, R * 32, 32, S, R, 1.0f, 0, 0);
    float* opmf = F32(B4);
    gemm(T, (long long)R * 32 * 32, 1024, w_opm_o, 0, CZ, 0, opmf, nullptr,
         (long long)R * CZ, CZ, nullptr, R, CZ, 1024, R, 1.0f, 0, 0);
    gemm(maskT_bf, 0, S, mask_bf, 0, R, 0, normf, nullptr, 0, R, nullptr,
         R, R, S, 1, 1.0f, 0, 0);
    evo_opm_add_kernel<<<blocks(PAIR_N), 256, 0, stream>>>(pair, opmf, in(38), normf, CZ, PAIR_N);
  }

  // ===== 5/6. Triangle multiplication outgoing / incoming =====
  for (int t = 0; t < 2; ++t) {
    int b0 = triBase[t];
    int swp = t; // outgoing: plain, incoming: swapped indices
    ln(pair, b0 + 0, b0 + 1, actb, nullptr, R * R, CZ);
    float* t1 = F32(B2);
    float* t2 = F32(B2 + 32 * MB);
    __bf16* TA = BFP(B3);
    __bf16* TB = BFP(B3 + 16 * MB);
    gemm(actb, 0, CZ, tri_lw[t], 0, CZ, 0, t1, nullptr, 0, CZ, in(b0 + 3), R * R, CZ, CZ, 1, 1.0f, 0, 0);
    gemm(actb, 0, CZ, tri_lgw[t], 0, CZ, 0, t2, nullptr, 0, CZ, in(b0 + 5), R * R, CZ, CZ, 1, 1.0f, 2, 0);
    evo_chan_ct_kernel<<<blocks(PAIR_N), 256, 0, stream>>>(t1, t2, in(3),
        (t == 0 ? TA : TB), R, CZ, swp, PAIR_N);
    gemm(actb, 0, CZ, tri_rw[t], 0, CZ, 0, t1, nullptr, 0, CZ, in(b0 + 7), R * R, CZ, CZ, 1, 1.0f, 0, 0);
    gemm(actb, 0, CZ, tri_rgw[t], 0, CZ, 0, t2, nullptr, 0, CZ, in(b0 + 9), R * R, CZ, CZ, 1, 1.0f, 2, 0);
    evo_chan_ct_kernel<<<blocks(PAIR_N), 256, 0, stream>>>(t1, t2, in(3),
        (t == 0 ? TB : TA), R, CZ, swp, PAIR_N);
    float* Ot = F32(B4);
    gemm(TA, (long long)R * R, R, TB, (long long)R * R, R, 1, Ot, nullptr,
         (long long)R * R, R, nullptr, R, R, R, CZ, 1.0f, 0, 0);
    float* of = F32(B2);
    evo_chan_ut_kernel<<<blocks(PAIR_N), 256, 0, stream>>>(Ot, of, R, CZ, PAIR_N);
    __bf16* cb = BFP(B3 + 32 * MB);
    ln(of, b0 + 10, b0 + 11, cb, nullptr, R * R, CZ);
    float* proj = F32(B4);
    gemm(cb, 0, CZ, tri_ow[t], 0, CZ, 0, proj, nullptr, 0, CZ, in(b0 + 13), R * R, CZ, CZ, 1, 1.0f, 0, 0);
    float* gate = F32(B2);
    gemm(actb, 0, CZ, tri_gw[t], 0, CZ, 0, gate, nullptr, 0, CZ, in(b0 + 15), R * R, CZ, CZ, 1, 1.0f, 2, 0);
    evo_mul_add_res_kernel<<<blocks(PAIR_N), 256, 0, stream>>>(pair, proj, gate, PAIR_N);
  }

  // ===== 7/8. Triangle attention (starting / ending node) =====
  for (int t = 0; t < 2; ++t) {
    int b0 = taBase[t];
    const float* src = pair;
    if (t == 1) {
      float* pairT = F32(B5);
      evo_transpose3_kernel<<<blocks(PAIR_N), 256, 0, stream>>>(pair, pairT, R, R, CZ, PAIR_N);
      src = pairT;
    }
    ln(src, b0 + 0, b0 + 1, actb, nullptr, R * R, CZ);
    float* nbp = F32(B1 + 16 * MB);
    gemm(actb, 0, CZ, ta_f[t], 0, HP, 0, nbp, nullptr, 0, HP, nullptr, R * R, HP, CZ, 1, 1.0f, 0, 0);
    gemm(actb, 0, CZ, ta_q[t], 0, HDP, 0, nullptr, qb, 0, HDP, nullptr, R * R, HDP, CZ, 1, QS, 0, 0);
    gemm(actb, 0, CZ, ta_k[t], 0, HDP, 0, nullptr, kb, 0, HDP, nullptr, R * R, HDP, CZ, 1, 1.0f, 0, 0);
    gemm(actb, 0, CZ, ta_v[t], 0, HDP, 0, nullptr, vb, 0, HDP, nullptr, R * R, HDP, CZ, 1, 1.0f, 0, 0);
    gemm(actb, 0, CZ, ta_g[t], 0, HDP, 0, nullptr, gateb, 0, HDP, in(b0 + 7), R * R, HDP, CZ, 1, 1.0f, 2, 0);
    if (t == 0) attn(qb, kb, vb, nbp, in(3), R, 1, wabuf, R, R, R, HP, HDP);
    else        attn(qb, kb, vb, nbp, in(3), 1, R, wabuf, R, R, R, HP, HDP);
    evo_mul_gate_kernel<<<blocks(PAIR_N), 256, 0, stream>>>(wabuf, gateb, gatedb, PAIR_N);
    if (t == 0) {
      gemm(gatedb, 0, HDP, ta_o[t], 0, CZ, 0, pair, nullptr, 0, CZ, in(b0 + 9),
           R * R, CZ, HDP, 1, 1.0f, 0, 1);
    } else {
      float* eout = F32(B5 + 32 * MB);
      gemm(gatedb, 0, HDP, ta_o[t], 0, CZ, 0, eout, nullptr, 0, CZ, in(b0 + 9),
           R * R, CZ, HDP, 1, 1.0f, 0, 0);
      evo_add_transpose_kernel<<<blocks(PAIR_N), 256, 0, stream>>>(pair, eout, R, R, CZ, PAIR_N);
    }
  }

  // ===== 9. Pair transition =====
  {
    ln(pair, 91, 92, actb, nullptr, R * R, CZ);
    __bf16* hid = BFP(B5);
    gemm(actb, 0, CZ, w_pt1, 0, 512, 0, nullptr, hid, 0, 512, in(94),
         R * R, 512, CZ, 1, 1.0f, 1, 0);
    gemm(hid, 0, 512, w_pt2, 0, CZ, 0, pair, nullptr, 0, CZ, in(96),
         R * R, CZ, 512, 1, 1.0f, 0, 1);
  }
}
